// GAT_5574867550288
// MI455X (gfx1250) — matrix operand order repaired and checked by reading, not compiled, not run on hardware
//
#include <hip/hip_runtime.h>

typedef __attribute__((ext_vector_type(16))) __bf16 v16bf;
typedef __attribute__((ext_vector_type(8)))  __bf16 v8bf;
typedef __attribute__((ext_vector_type(8)))  float  v8f;

#define DEVINL __device__ __forceinline__

// monotonic unsigned encoding of float for atomicMax-based segment_max
DEVINL unsigned fenc(float f) {
    unsigned b = __float_as_uint(f);
    return (b & 0x80000000u) ? ~b : (b | 0x80000000u);
}
DEVINL float fdec(unsigned u) {
    return (u & 0x80000000u) ? __uint_as_float(u ^ 0x80000000u)
                             : __uint_as_float(~u);
}

__global__ void k_fill_f32(float* __restrict__ p, float v, int n) {
    int i = blockIdx.x * blockDim.x + threadIdx.x;
    if (i < n) p[i] = v;
}

__global__ void k_fill_u32(unsigned* __restrict__ p, unsigned v, int n) {
    int i = blockIdx.x * blockDim.x + threadIdx.x;
    if (i < n) p[i] = v;
}

__global__ void k_cvt_bf16(const float* __restrict__ x, __bf16* __restrict__ y, int n) {
    int i = blockIdx.x * blockDim.x + threadIdx.x;
    if (i < n) y[i] = (__bf16)x[i];
}

// Convert W [K,512] f32 -> bf16 packed into WMMA B-fragment order:
// fragment index = ((cg*KT + kt)*4 + t), per-lane contiguous 16 elements.
// B 32x16 layout used here (contiguous k, same order as the A load): lane = half*16 + (n%16); half = (k%32)/16; elem e = k%16.
__global__ void k_pack_w(const float* __restrict__ W, __bf16* __restrict__ Wp, int K) {
    int i = blockIdx.x * blockDim.x + threadIdx.x;
    int total = K * 512;
    if (i >= total) return;
    int k = i / 512, n = i % 512;
    int cg = n >> 6;          // 64-col group (0..7)
    int t  = (n >> 4) & 3;    // 16-col tile within group
    int l16 = n & 15;
    int kt  = k >> 5;         // 32-K tile
    int kin = k & 31;
    int half = kin >> 4;
    int e    = kin & 15;
    int lane = half * 16 + l16;
    int KT = K >> 5;
    Wp[((size_t)(((cg * KT + kt) * 4 + t) * 32 + lane) << 4) + e] = (__bf16)W[i];
}

// Y[N,512] = X[N,K] @ W[K,512], bf16 inputs, f32 accumulate via v_wmma.
// One wave per 16x64 output tile. grid = (N/16, 8), block = 32.
__global__ void __launch_bounds__(32)
k_gemm_wmma(const __bf16* __restrict__ X, const __bf16* __restrict__ Wp,
            float* __restrict__ Y, int K) {
    const int lane = threadIdx.x;
    const int half = lane >> 4, l16 = lane & 15;
    const int m0 = blockIdx.x * 16;
    const int cg = blockIdx.y;
    const int KT = K >> 5;

    v8f acc[4] = {};

    for (int kt = 0; kt < KT; ++kt) {
        // A 16x32 bf16: half 0 -> K {0..15}, half 1 -> K {16..31}
        const __bf16* xr = X + (size_t)(m0 + l16) * K + kt * 32 + half * 16;
        v8bf lo = *(const v8bf*)xr;
        v8bf hi = *(const v8bf*)(xr + 8);
        v16bf a = __builtin_shufflevector(lo, hi, 0, 1, 2, 3, 4, 5, 6, 7,
                                          8, 9, 10, 11, 12, 13, 14, 15);
        const __bf16* wb = Wp + ((size_t)((cg * KT + kt) * 4) * 32 + lane) * 16;
#pragma unroll
        for (int t = 0; t < 4; ++t) {
            v16bf b = *(const v16bf*)(wb + (size_t)t * 512);
            acc[t] = __builtin_amdgcn_wmma_f32_16x16x32_bf16(
                false, a, false, b, (short)0, acc[t], false, false);
        }
    }
    // C/D layout: VGPR r -> row m0 + r + 8*half, col = l16 within tile
#pragma unroll
    for (int t = 0; t < 4; ++t) {
        float* yr = Y + (size_t)(m0 + 8 * half) * 512 + cg * 64 + t * 16 + l16;
#pragma unroll
        for (int r = 0; r < 8; ++r) yr[(size_t)r * 512] = acc[t][r];
    }
}

// a_s[n,h] = sum_c xs[n,h,c] * as[h,c]
__global__ void k_attn_src(const float* __restrict__ xs, const float* __restrict__ av,
                           float* __restrict__ out, int N) {
    int i = blockIdx.x * blockDim.x + threadIdx.x;
    if (i >= N * 8) return;
    int n = i >> 3, h = i & 7;
    const float* xr = xs + (size_t)n * 512 + h * 64;
    const float* ar = av + h * 64;
    float s = 0.f;
    for (int c = 0; c < 64; ++c) s += xr[c] * ar[c];
    out[i] = s;
}

// wda[k,h] = sum_c Wd[k, h*64+c] * ad[h,c]   (single block, K*8 threads)
__global__ void k_wda(const float* __restrict__ Wd, const float* __restrict__ ad,
                      float* __restrict__ out, int K) {
    int i = threadIdx.x;
    if (i >= K * 8) return;
    int k = i >> 3, h = i & 7;
    const float* wr = Wd + (size_t)k * 512 + h * 64;
    const float* ar = ad + h * 64;
    float s = 0.f;
    for (int c = 0; c < 64; ++c) s += wr[c] * ar[c];
    out[i] = s;
}

// a_d[n,h] = sum_k x[n,k] * wda[k,h]
__global__ void k_attn_dst(const float* __restrict__ x, const float* __restrict__ wda,
                           float* __restrict__ out, int N, int K) {
    int i = blockIdx.x * blockDim.x + threadIdx.x;
    if (i >= N * 8) return;
    int n = i >> 3, h = i & 7;
    const float* xr = x + (size_t)n * K;
    float s = 0.f;
    for (int k = 0; k < K; ++k) s += xr[k] * wda[k * 8 + h];
    out[i] = s;
}

__global__ void k_edge_max(const int* __restrict__ src, const int* __restrict__ dst,
                           const float* __restrict__ as_, const float* __restrict__ ad_,
                           unsigned* __restrict__ menc, int E) {
    int i = blockIdx.x * blockDim.x + threadIdx.x;
    if (i >= E * 8) return;
    int e = i >> 3, h = i & 7;
    float v = as_[src[e] * 8 + h] + ad_[dst[e] * 8 + h];
    v = v > 0.f ? v : 0.2f * v;                    // leaky_relu 0.2
    atomicMax(&menc[dst[e] * 8 + h], fenc(v));
}

__global__ void k_edge_exp(const int* __restrict__ src, const int* __restrict__ dst,
                           const float* __restrict__ as_, const float* __restrict__ ad_,
                           const unsigned* __restrict__ menc, float* __restrict__ z,
                           float* __restrict__ ebuf, int E) {
    int i = blockIdx.x * blockDim.x + threadIdx.x;
    if (i >= E * 8) return;
    int e = i >> 3, h = i & 7;
    int d8 = dst[e] * 8 + h;
    float v = as_[src[e] * 8 + h] + ad_[d8];
    v = v > 0.f ? v : 0.2f * v;
    float ex = __expf(v - fdec(menc[d8]));
    ebuf[i] = ex;
    unsafeAtomicAdd(&z[d8], ex);
}

// one wave per edge: acc[dst,h,c] += xs[src,h,c] * alpha[e,h]
__global__ void __launch_bounds__(256)
k_edge_msg(const int* __restrict__ src, const int* __restrict__ dst,
           const float* __restrict__ xs, const float* __restrict__ ebuf,
           const float* __restrict__ z, float* __restrict__ acc, int E) {
    int gw = (int)((blockIdx.x * 256u + threadIdx.x) >> 5);
    int lane = threadIdx.x & 31;
    if (gw >= E) return;
    int s = src[gw], d = dst[gw];
    float al[8];
#pragma unroll
    for (int h = 0; h < 8; ++h)
        al[h] = ebuf[(size_t)gw * 8 + h] / (z[d * 8 + h] + 1e-16f);
    const float* xr = xs + (size_t)s * 512;
    float* ar = acc + (size_t)d * 512;
#pragma unroll
    for (int j = 0; j < 16; ++j) {
        int c = lane + 32 * j;
        unsafeAtomicAdd(&ar[c], xr[c] * al[c >> 6]);
    }
}

// x_out[n,c] = ELU( mean_h acc[n,h,c] + sum of relation biases )
__global__ void k_finalize(const float* __restrict__ acc,
                           const float* b0, const float* b1,
                           const float* b2, const float* b3,
                           float* __restrict__ xout, int N) {
    int i = blockIdx.x * blockDim.x + threadIdx.x;
    if (i >= N * 64) return;
    int n = i >> 6, c = i & 63;
    float s = 0.f;
#pragma unroll
    for (int h = 0; h < 8; ++h) s += acc[(size_t)n * 512 + h * 64 + c];
    s *= 0.125f;
    if (b0) s += b0[c];
    if (b1) s += b1[c];
    if (b2) s += b2[c];
    if (b3) s += b3[c];
    xout[i] = s > 0.f ? s : expm1f(s);             // ELU(alpha=1)
}

__global__ void k_pool(const float* __restrict__ x, const int* __restrict__ batch,
                       float* __restrict__ psum, float* __restrict__ cnt,
                       int N, int toff) {
    int i = blockIdx.x * blockDim.x + threadIdx.x;
    if (i >= N * 64) return;
    int n = i >> 6, c = i & 63;
    int b = batch[n];
    unsafeAtomicAdd(&psum[b * 192 + toff + c], x[i]);
    if (c == 0) unsafeAtomicAdd(&cnt[b], 1.0f);
}

__global__ void __launch_bounds__(64)
k_head(const float* __restrict__ psum, const float* __restrict__ cnt,
       const float* __restrict__ lw, const float* __restrict__ lb,
       float* __restrict__ out) {
    int b = threadIdx.x;   // 0..63 graphs
    float lg[10];
#pragma unroll
    for (int o = 0; o < 10; ++o) lg[o] = lb[o];
    for (int t = 0; t < 3; ++t) {
        float ct = cnt[t * 64 + b];
        ct = ct > 1.f ? ct : 1.f;
        float inv = 1.f / ct;
        for (int i = 0; i < 64; ++i) {
            float hv = psum[b * 192 + t * 64 + i] * inv;
            const float* wr = lw + (t * 64 + i) * 10;
#pragma unroll
            for (int o = 0; o < 10; ++o) lg[o] += hv * wr[o];
        }
    }
    float mx = lg[0];
#pragma unroll
    for (int o = 1; o < 10; ++o) mx = fmaxf(mx, lg[o]);
    float s = 0.f;
#pragma unroll
    for (int o = 0; o < 10; ++o) s += __expf(lg[o] - mx);
    float lse = mx + logf(s);
#pragma unroll
    for (int o = 0; o < 10; ++o) out[b * 10 + o] = lg[o] - lse;
}

extern "C" void kernel_launch(void* const* d_in, const int* in_sizes, int n_in,
                              void* d_out, int out_size, void* d_ws, size_t ws_size,
                              hipStream_t stream) {
    (void)in_sizes; (void)n_in; (void)out_size; (void)ws_size;

    const int NUMT[3] = {40000, 20000, 10000};      // control, variable, constant
    struct RelT { int s, d, E; };
    const RelT rels[6] = {{0,0,300000},{0,0,150000},{0,1,200000},
                          {1,0,200000},{2,0,100000},{0,2,100000}};

    // params pytree leaf order: layers[L][rj]{Wd,Ws,ad,as,b}, then lin_b, lin_w
    auto P = [&](int L, int j, int which) -> const float* {
        return (const float*)d_in[12 + (L * 6 + j) * 5 + which];
    };

    char* ws = (char*)d_ws;
    size_t off = 0;
    auto alloc = [&](size_t bytes) -> char* {
        char* p = ws + off;
        off = (off + bytes + 255) & ~(size_t)255;
        return p;
    };

    float* acc[3];
    for (int t = 0; t < 3; ++t) acc[t] = (float*)alloc((size_t)NUMT[t] * 512 * 4);
    float*  xs = (float*)alloc((size_t)40000 * 512 * 4);
    __bf16* xb[3];
    for (int t = 0; t < 3; ++t) xb[t] = (__bf16*)alloc((size_t)NUMT[t] * 128 * 2);
    float*  xf[3];
    for (int t = 0; t < 3; ++t) xf[t] = (float*)alloc((size_t)NUMT[t] * 64 * 4);
    __bf16* Wp   = (__bf16*)alloc((size_t)128 * 512 * 2);
    float*  wda  = (float*)alloc(128 * 8 * 4);
    float*  a_s  = (float*)alloc((size_t)40000 * 8 * 4);
    float*  a_d  = (float*)alloc((size_t)40000 * 8 * 4);
    unsigned* menc = (unsigned*)alloc((size_t)40000 * 8 * 4);
    float*  zbuf = (float*)alloc((size_t)40000 * 8 * 4);
    float*  ebuf = (float*)alloc((size_t)300000 * 8 * 4);
    float*  psum = (float*)alloc(64 * 192 * 4);
    float*  cnt  = (float*)alloc(192 * 4);

    auto g = [](int n, int b) { return (n + b - 1) / b; };

    const float* xin[3];
    for (int L = 0; L < 2; ++L) {
        int K = L ? 64 : 128;
        for (int t = 0; t < 3; ++t) xin[t] = L ? xf[t] : (const float*)d_in[t];

        for (int t = 0; t < 3; ++t) {
            int na = NUMT[t] * 512;
            k_fill_f32<<<g(na, 256), 256, 0, stream>>>(acc[t], 0.f, na);
            int nx = NUMT[t] * K;
            k_cvt_bf16<<<g(nx, 256), 256, 0, stream>>>(xin[t], xb[t], nx);
        }

        for (int j = 0; j < 6; ++j) {
            int sT = rels[j].s, dT = rels[j].d, E = rels[j].E;
            int Ns = NUMT[sT], Nd = NUMT[dT];
            const float* Wd  = P(L, j, 0);
            const float* Ws  = P(L, j, 1);
            const float* ad_ = P(L, j, 2);
            const float* as_ = P(L, j, 3);
            const int* ei   = (const int*)d_in[3 + j];
            const int* esrc = ei;
            const int* edst = ei + E;

            k_pack_w<<<g(K * 512, 256), 256, 0, stream>>>(Ws, Wp, K);
            k_gemm_wmma<<<dim3(Ns / 16, 8), 32, 0, stream>>>(xb[sT], Wp, xs, K);
            k_attn_src<<<g(Ns * 8, 256), 256, 0, stream>>>(xs, as_, a_s, Ns);
            k_wda<<<1, K * 8, 0, stream>>>(Wd, ad_, wda, K);
            k_attn_dst<<<g(Nd * 8, 256), 256, 0, stream>>>(xin[dT], wda, a_d, Nd, K);
            k_fill_u32<<<g(Nd * 8, 256), 256, 0, stream>>>(menc, 0x007FFFFFu, Nd * 8);
            k_fill_f32<<<g(Nd * 8, 256), 256, 0, stream>>>(zbuf, 0.f, Nd * 8);
            k_edge_max<<<g(E * 8, 256), 256, 0, stream>>>(esrc, edst, a_s, a_d, menc, E);
            k_edge_exp<<<g(E * 8, 256), 256, 0, stream>>>(esrc, edst, a_s, a_d, menc, zbuf, ebuf, E);
            k_edge_msg<<<g(E * 32, 256), 256, 0, stream>>>(esrc, edst, xs, ebuf, zbuf, acc[dT], E);
        }

        for (int t = 0; t < 3; ++t) {
            const float *b0 = nullptr, *b1 = nullptr, *b2 = nullptr, *b3 = nullptr;
            if (t == 0) { b0 = P(L,0,4); b1 = P(L,1,4); b2 = P(L,3,4); b3 = P(L,4,4); }
            else if (t == 1) { b0 = P(L,2,4); }
            else { b0 = P(L,5,4); }
            k_finalize<<<g(NUMT[t] * 64, 256), 256, 0, stream>>>(acc[t], b0, b1, b2, b3, xf[t], NUMT[t]);
        }
    }

    k_fill_f32<<<g(64 * 192, 256), 256, 0, stream>>>(psum, 0.f, 64 * 192);
    k_fill_f32<<<1, 192, 0, stream>>>(cnt, 0.f, 192);
    for (int t = 0; t < 3; ++t)
        k_pool<<<g(NUMT[t] * 64, 256), 256, 0, stream>>>(
            xf[t], (const int*)d_in[9 + t], psum, cnt + t * 64, NUMT[t], t * 64);
    k_head<<<1, 64, 0, stream>>>(psum, cnt, (const float*)d_in[73],
                                 (const float*)d_in[72], (float*)d_out);
}
